// WindowAttention_7060926234822
// MI455X (gfx1250) — matrix operand order repaired and checked by reading, not compiled, not run on hardware
//
#include <hip/hip_runtime.h>

// ---------------------------------------------------------------------------
// Fused 8x8 window attention for MI455X (gfx1250, wave32, WMMA bf16).
// One workgroup (256 thr = 8 waves) per window; wave w == head w.
// All matmuls via v_wmma_f32_16x16x32_bf16; x read once, out written once.
// ---------------------------------------------------------------------------

#define NHEADS 8
#define TOK    64      // 8x8 tokens per window
#define CH     256
#define HD     32      // head dim

typedef __attribute__((ext_vector_type(16))) __bf16 bf16x16;
typedef __attribute__((ext_vector_type(8)))  __bf16 bf16x8;
typedef __attribute__((ext_vector_type(4)))  __bf16 bf16x4;
typedef __attribute__((ext_vector_type(8)))  float  f32x8;

// LDS geometry (elements). All row strides are multiples of 8 elems (16B)
// so every b128 ds access stays 16-byte aligned.
#define XS_STRIDE 264                       // x tile / o tile: [64][256+8]
#define QT_STRIDE 72                        // qT / vT: [32][64+8]
#define KS_STRIDE 40                        // k row-major: [64][32+8]
#define PT_STRIDE 72                        // pT: [64][64+8]
#define XS_BYTES  (64 * XS_STRIDE * 2)      // 33792
#define QT_BYTES  (32 * QT_STRIDE * 2)      // 4608
#define KS_BYTES  (64 * KS_STRIDE * 2)      // 5120
#define PT_BYTES  (64 * PT_STRIDE * 2)      // 9216
#define HEAD_BYTES (2 * QT_BYTES + KS_BYTES + PT_BYTES)   // 23552
#define SMEM_BYTES (XS_BYTES + NHEADS * HEAD_BYTES)       // 222208 (< 320KB/WGP)

union FragU { bf16x16 v; bf16x8 h[2]; };

// A-frag (16x32 bf16): lane L -> row M; elems 0..7 = K kb+half*8.., 8..15 = K kb+16+half*8..
__device__ __forceinline__ bf16x16 load_a(const __bf16* base, int stride,
                                          int row, int kb, int half) {
  FragU f;
  const __bf16* p = base + row * stride + kb + half * 8;
  f.h[0] = *(const bf16x8*)p;
  f.h[1] = *(const bf16x8*)(p + 16);
  return f.v;
}

// B-frag (32x16 bf16): lane L -> col N = nb+L%16; elems 0..7 = K kb+half*8.., 8..15 = K kb+16+half*8.. (row-major [K][N]).
__device__ __forceinline__ bf16x16 load_b(const __bf16* base, int stride,
                                          int kb, int nb, int lane) {
  FragU f;
  const __bf16* p = base + (kb + (lane >> 4) * 8) * stride + nb + (lane & 15);
#pragma unroll
  for (int s = 0; s < 8; ++s) {
    f.v[s] = p[s * stride];
    f.v[8 + s] = p[(16 + s) * stride];
  }
  return f.v;
}

__device__ __forceinline__ f32x8 wmma_bf16(bf16x16 a, bf16x16 b, f32x8 c) {
  return __builtin_amdgcn_wmma_f32_16x16x32_bf16(false, a, false, b,
                                                 (short)0, c, false, false);
}

__device__ __forceinline__ bf16x8 cvt8(f32x8 a, float bias) {
  bf16x8 r;
#pragma unroll
  for (int i = 0; i < 8; ++i) r[i] = (__bf16)(a[i] + bias);
  return r;
}

__global__ void __launch_bounds__(256)
cvt_f32_to_bf16(const float* __restrict__ src, __bf16* __restrict__ dst, int n) {
  int i = blockIdx.x * blockDim.x + threadIdx.x;
  if (i < n) dst[i] = (__bf16)src[i];
}

__global__ void __launch_bounds__(256)
swin_window_attn(const float* __restrict__ x,
                 const float* __restrict__ b_qkv,
                 const float* __restrict__ b_proj,
                 const __bf16* __restrict__ wqkv,   // [256][768] bf16
                 const __bf16* __restrict__ wproj,  // [256][256] bf16
                 float* __restrict__ out) {
  extern __shared__ char smem[];
  const int tid  = threadIdx.x;
  const int lane = tid & 31;
  const int head = tid >> 5;
  const int l16  = lane & 15;
  const int half = lane >> 4;

  const int win = blockIdx.x;          // 4096 windows
  const int bb  = win >> 10;
  const int rem = win & 1023;
  const int wy  = rem >> 5;
  const int wx  = rem & 31;

  __bf16* xs = (__bf16*)smem;          // x tile; later reused as O tile
  char* hb = smem + XS_BYTES + head * HEAD_BYTES;
  __bf16* qsT = (__bf16*)hb;                              // [32 c][72]  q^T
  __bf16* vsT = (__bf16*)(hb + QT_BYTES);                 // [32 d][72]  v^T
  __bf16* ksm = (__bf16*)(hb + 2 * QT_BYTES);             // [64 s][40]  k row-major
  __bf16* psT = (__bf16*)(hb + 2 * QT_BYTES + KS_BYTES);  // [64 s][72]  P^T

  // gfx1250 speculative prefetch of the weight stream (global_prefetch_b8)
  __builtin_prefetch(wqkv + (size_t)tid * 768, 0, 0);

  // ---- stage x window -> LDS bf16 (read x exactly once) ----
  {
    const int cq = (tid & 63) << 2;            // channel quad
#pragma unroll
    for (int i = 0; i < 16; ++i) {
      const int t  = (i << 2) + (tid >> 6);
      const int yy = (wy << 3) + (t >> 3);
      const int xx = (wx << 3) + (t & 7);
      const float4 f =
          *(const float4*)(x + ((size_t)(((bb << 8) + yy) * 256 + xx) << 8) + cq);
      bf16x4 h;
      h[0] = (__bf16)f.x; h[1] = (__bf16)f.y;
      h[2] = (__bf16)f.z; h[3] = (__bf16)f.w;
      *(bf16x4*)(xs + t * XS_STRIDE + cq) = h;
    }
  }
  __syncthreads();

  const f32x8 z8 = {0.f, 0.f, 0.f, 0.f, 0.f, 0.f, 0.f, 0.f};

  // ---- QKV GEMM: [64,256] x [256,32] per matrix, per head ----
#pragma unroll
  for (int m = 0; m < 3; ++m) {                // 0=q 1=k 2=v
    f32x8 acc[4][2];
#pragma unroll
    for (int mt = 0; mt < 4; ++mt)
#pragma unroll
      for (int nt = 0; nt < 2; ++nt) acc[mt][nt] = z8;

#pragma unroll
    for (int kk = 0; kk < 8; ++kk) {
      const int kb = kk << 5;
      bf16x16 a[4];
#pragma unroll
      for (int mt = 0; mt < 4; ++mt)
        a[mt] = load_a(xs, XS_STRIDE, mt * 16 + l16, kb, half);
#pragma unroll
      for (int nt = 0; nt < 2; ++nt) {
        bf16x16 b = load_b(wqkv, 768, kb, m * 256 + head * 32 + nt * 16, lane);
#pragma unroll
        for (int mt = 0; mt < 4; ++mt)
          acc[mt][nt] = wmma_bf16(a[mt], b, acc[mt][nt]);
      }
    }
    // C-layout: elem e -> row mt*16+half*8+e, col nt*16+l16
#pragma unroll
    for (int nt = 0; nt < 2; ++nt) {
      const float bias = b_qkv[m * 256 + head * 32 + nt * 16 + l16];
#pragma unroll
      for (int mt = 0; mt < 4; ++mt) {
        if (m == 1) {  // k row-major [s][c]: strided b16 stores
#pragma unroll
          for (int e = 0; e < 8; ++e) {
            const int s = mt * 16 + half * 8 + e;
            ksm[s * KS_STRIDE + nt * 16 + l16] = (__bf16)(acc[mt][nt][e] + bias);
          }
        } else {       // q^T / v^T: contiguous b128 stores
          __bf16* dst = (m == 0) ? qsT : vsT;
          *(bf16x8*)(dst + (nt * 16 + l16) * QT_STRIDE + mt * 16 + half * 8) =
              cvt8(acc[mt][nt], bias);
        }
      }
    }
  }
  __syncthreads();   // xs now dead; q/k/v ready

  // ---- S^T = K @ Q^T  (64x64, K=32 -> one WMMA step) ----
  f32x8 st[4][4];
#pragma unroll
  for (int mt = 0; mt < 4; ++mt)
#pragma unroll
    for (int nt = 0; nt < 4; ++nt) st[mt][nt] = z8;
  {
    bf16x16 a[4];
#pragma unroll
    for (int mt = 0; mt < 4; ++mt)
      a[mt] = load_a(ksm, KS_STRIDE, mt * 16 + l16, 0, half);
#pragma unroll
    for (int nt = 0; nt < 4; ++nt) {
      bf16x16 b = load_b(qsT, QT_STRIDE, 0, nt * 16, lane);
#pragma unroll
      for (int mt = 0; mt < 4; ++mt)
        st[mt][nt] = wmma_bf16(a[mt], b, st[mt][nt]);
    }
  }

  // ---- softmax over s (per token t); lanes l and l^16 hold the same t ----
  const float scale = 0.17677669529663687f;  // 1/sqrt(32)
#pragma unroll
  for (int nt = 0; nt < 4; ++nt) {
    float mx = -3.0e38f;
#pragma unroll
    for (int mt = 0; mt < 4; ++mt)
#pragma unroll
      for (int e = 0; e < 8; ++e) {
        st[mt][nt][e] *= scale;
        mx = fmaxf(mx, st[mt][nt][e]);
      }
    mx = fmaxf(mx, __shfl_xor(mx, 16, 32));
    float sum = 0.f;
#pragma unroll
    for (int mt = 0; mt < 4; ++mt)
#pragma unroll
      for (int e = 0; e < 8; ++e) {
        const float p = __expf(st[mt][nt][e] - mx);
        st[mt][nt][e] = p;
        sum += p;
      }
    sum += __shfl_xor(sum, 16, 32);
    const float inv = 1.0f / sum;
#pragma unroll
    for (int mt = 0; mt < 4; ++mt)
#pragma unroll
      for (int e = 0; e < 8; ++e) {
        const int s = mt * 16 + half * 8 + e;
        psT[s * PT_STRIDE + nt * 16 + l16] = (__bf16)(st[mt][nt][e] * inv);
      }
  }
  __syncthreads();

  // ---- O^T = V^T @ P^T  (32x64, K=64) -> lands contiguously into o tile ----
  {
    f32x8 ot[2][4];
#pragma unroll
    for (int mt = 0; mt < 2; ++mt)
#pragma unroll
      for (int nt = 0; nt < 4; ++nt) ot[mt][nt] = z8;
#pragma unroll
    for (int kk = 0; kk < 2; ++kk) {
      const int kb = kk << 5;
      bf16x16 a[2];
#pragma unroll
      for (int mt = 0; mt < 2; ++mt)
        a[mt] = load_a(vsT, QT_STRIDE, mt * 16 + l16, kb, half);
#pragma unroll
      for (int nt = 0; nt < 4; ++nt) {
        bf16x16 b = load_b(psT, PT_STRIDE, kb, nt * 16, lane);
#pragma unroll
        for (int mt = 0; mt < 2; ++mt)
          ot[mt][nt] = wmma_bf16(a[mt], b, ot[mt][nt]);
      }
    }
    // C-layout of O^T: lane -> token t, elems -> 8 consecutive d: b128 stores
#pragma unroll
    for (int nt = 0; nt < 4; ++nt) {
      const int t = nt * 16 + l16;
#pragma unroll
      for (int mt = 0; mt < 2; ++mt)
        *(bf16x8*)(xs + t * XS_STRIDE + head * 32 + mt * 16 + half * 8) =
            cvt8(ot[mt][nt], 0.f);
    }
  }
  __syncthreads();   // full [64][256] O tile visible to all waves

  // ---- projection: [64,256] x [256,32] per head; write out once ----
  {
    f32x8 po[4][2];
#pragma unroll
    for (int mt = 0; mt < 4; ++mt)
#pragma unroll
      for (int nt = 0; nt < 2; ++nt) po[mt][nt] = z8;
#pragma unroll
    for (int kk = 0; kk < 8; ++kk) {
      const int kb = kk << 5;
      bf16x16 a[4];
#pragma unroll
      for (int mt = 0; mt < 4; ++mt)
        a[mt] = load_a(xs, XS_STRIDE, mt * 16 + l16, kb, half);
#pragma unroll
      for (int nt = 0; nt < 2; ++nt) {
        bf16x16 b = load_b(wproj, 256, kb, head * 32 + nt * 16, lane);
#pragma unroll
        for (int mt = 0; mt < 4; ++mt)
          po[mt][nt] = wmma_bf16(a[mt], b, po[mt][nt]);
      }
    }
#pragma unroll
    for (int nt = 0; nt < 2; ++nt) {
      const int n = head * 32 + nt * 16 + l16;
      const float bias = b_proj[n];
#pragma unroll
      for (int mt = 0; mt < 4; ++mt)
#pragma unroll
        for (int e = 0; e < 8; ++e) {
          const int t  = mt * 16 + half * 8 + e;
          const int yy = (wy << 3) + (t >> 3);
          const int xx = (wx << 3) + (t & 7);
          out[((size_t)(((bb << 8) + yy) * 256 + xx) << 8) + n] =
              po[mt][nt][e] + bias;
        }
    }
  }
}

extern "C" void kernel_launch(void* const* d_in, const int* in_sizes, int n_in,
                              void* d_out, int out_size, void* d_ws, size_t ws_size,
                              hipStream_t stream) {
  (void)in_sizes; (void)n_in; (void)out_size; (void)ws_size;
  const float* x      = (const float*)d_in[0];
  const float* w_qkv  = (const float*)d_in[1];
  const float* b_qkv  = (const float*)d_in[2];
  const float* w_proj = (const float*)d_in[3];
  const float* b_proj = (const float*)d_in[4];

  // scratch: bf16 copies of the (tiny, L2-resident) weights
  __bf16* wqkv_bf  = (__bf16*)d_ws;                   // 256*768 = 196608 elems
  __bf16* wproj_bf = wqkv_bf + 256 * 768;             // 256*256 =  65536 elems

  cvt_f32_to_bf16<<<(196608 + 255) / 256, 256, 0, stream>>>(w_qkv, wqkv_bf, 196608);
  cvt_f32_to_bf16<<<(65536 + 255) / 256, 256, 0, stream>>>(w_proj, wproj_bf, 65536);

  // 4 * 32 * 32 = 4096 windows, one workgroup each, 8 waves = 8 heads
  swin_window_attn<<<4096, 256, SMEM_BYTES, stream>>>(
      x, b_qkv, b_proj, wqkv_bf, wproj_bf, (float*)d_out);
}
